// Node_2989297238563
// MI455X (gfx1250) — matrix operand order repaired and checked by reading, not compiled, not run on hardware
//
#include <hip/hip_runtime.h>
#include <hip/hip_bf16.h>
#include <math.h>

typedef __attribute__((ext_vector_type(16))) _Float16     v16h;
typedef __attribute__((ext_vector_type(2)))  __fp16       v2fp;  // type returned by cvt_pkrtz
typedef __attribute__((ext_vector_type(8)))  float        v8f;
typedef __attribute__((ext_vector_type(4)))  unsigned int u32x4;
typedef __attribute__((ext_vector_type(2)))  unsigned int u32x2;
typedef __attribute__((ext_vector_type(4)))  float        f32x4;

#define B_ROWS   65536
#define D_IN     1024
#define N_NODES  256      // padded; 255 real internal nodes
#define N_REAL   255
#define D_OUT    128
#define M_TILE   128
#define K_CHUNK  32
#define BLOCK    256
#define N_BLOCKS (B_ROWS / M_TILE)   // 512

union FragU { u32x4 q[2]; v16h h; };

// A-fragment (16x32 f16, MxK): lanes 0-15 = rows, half selects K {0..7,16..23} vs {8..15,24..31}
__device__ inline v16h load_frag_a(const _Float16* base, int stride, int row16, int k0, int lane) {
  int m = lane & 15, half = lane >> 4;
  const _Float16* p = base + (size_t)(row16 + m) * stride + k0 + 8 * half;
  FragU u;
  u.q[0] = *(const u32x4*)(p);
  u.q[1] = *(const u32x4*)(p + 16);
  return u.h;
}
// B-fragment (32x16 f16, KxN) from N-major storage: half selects K {0..7,16..23} vs {8..15,24..31}
__device__ inline v16h load_frag_b(const _Float16* base, int stride, int col16, int k0, int lane) {
  int n = lane & 15, half = lane >> 4;
  const _Float16* p = base + (size_t)(col16 + n) * stride + k0 + 8 * half;
  FragU u;
  u.q[0] = *(const u32x4*)(p);
  u.q[1] = *(const u32x4*)(p + 16);
  return u.h;
}

__device__ inline unsigned lds_offset_of(const void* p) {
  // generic pointers into LDS carry the AS(3) byte offset in their low 32 bits
  return (unsigned)(unsigned long long)p;
}

// ---- prep: Wt[n][k] = W_split[k][n] (f16, n=255 zero-padded) ----
__global__ void prep_wt(const float* __restrict__ W, _Float16* __restrict__ Wt) {
  int idx = blockIdx.x * blockDim.x + threadIdx.x;   // 0..262143
  int n = idx >> 10, k = idx & 1023;
  float v = (n < N_REAL) ? W[(size_t)k * N_REAL + n] : 0.0f;
  Wt[idx] = (_Float16)v;
}
// ---- prep: leafT[n][k] = leaf_values[k][n] (f16) ----
__global__ void prep_leaf(const float* __restrict__ L, _Float16* __restrict__ leafT) {
  int idx = blockIdx.x * blockDim.x + threadIdx.x;   // 0..32767
  int n = idx >> 8, k = idx & 255;
  leafT[idx] = (_Float16)L[(size_t)k * D_OUT + n];
}

__global__ __launch_bounds__(BLOCK) void fused_tree_kernel(
    const float* __restrict__ x, const float* __restrict__ b_split,
    const _Float16* __restrict__ Wt, const _Float16* __restrict__ leafT,
    float* __restrict__ out, float* __restrict__ regPartial)
{
  __shared__ char smem[131072];
  // GEMM1 double-buffered tiles (scalar pointers only; no pointer arrays!)
  _Float16* aBuf0 = (_Float16*)smem;             // [128][32] 8KB
  _Float16* aBuf1 = (_Float16*)(smem + 8192);    // [128][32] 8KB
  _Float16* bBuf0 = (_Float16*)(smem + 16384);   // [256][32] 16KB
  _Float16* bBuf1 = (_Float16*)(smem + 49152);   // [256][32] 16KB
  // post-GEMM1 reuse:
  _Float16* bufP = (_Float16*)smem;              // [128][256] 64KB  sigmoid probs
  _Float16* bufQ = (_Float16*)(smem + 65536);    // [128][256] 64KB  leaf probs

  const int tid  = threadIdx.x;
  const int lane = tid & 31, wave = tid >> 5;
  const int half = lane >> 4, l16 = lane & 15;
  const int rowBase = blockIdx.x * M_TILE;

  const unsigned bOff0 = lds_offset_of(bBuf0);
  const unsigned bOff1 = lds_offset_of(bBuf1);
  const unsigned long long wtBase = (unsigned long long)Wt;

  // ================= GEMM1: logits[128][256] = x_tile @ Wt^T =================
  const int waveM = wave >> 2;   // 0..1 -> 64-row band
  const int waveN = wave & 3;    // 0..3 -> 64-col band
  v8f vzero = {0.f,0.f,0.f,0.f,0.f,0.f,0.f,0.f};
  v8f acc[4][4];
  for (int mt = 0; mt < 4; ++mt)
    for (int nt = 0; nt < 4; ++nt) acc[mt][nt] = vzero;

  f32x4 pre[4];   // register prefetch of next x chunk
  auto loadA = [&](int k0) {
    for (int it = 0; it < 4; ++it) {
      int e4 = (tid + it * BLOCK) * 4;          // 0..4092
      int r = e4 >> 5, kk = e4 & 31;
      pre[it] = *(const f32x4*)(x + (size_t)(rowBase + r) * D_IN + k0 + kk);
    }
  };
  auto storeA = [&](_Float16* dstBuf) {
    for (int it = 0; it < 4; ++it) {
      int e4 = (tid + it * BLOCK) * 4;
      int r = e4 >> 5, kk = e4 & 31;
      union { v2fp h[2]; u32x2 u; } cv;
      cv.h[0] = __builtin_amdgcn_cvt_pkrtz(pre[it].x, pre[it].y);
      cv.h[1] = __builtin_amdgcn_cvt_pkrtz(pre[it].z, pre[it].w);
      *(u32x2*)(dstBuf + r * 32 + kk) = cv.u;
    }
  };
  // async DMA of the (already f16) Wt tile straight into LDS, one chunk ahead
  auto asyncB = [&](unsigned dstOff, int k0) {
    unsigned ldsOff = dstOff + (unsigned)tid * 64u;               // 64B per thread (row tid)
    unsigned gOff   = ((unsigned)tid * D_IN + (unsigned)k0) * 2u; // byte offset from Wt
    for (int j = 0; j < 4; ++j)
      asm volatile("global_load_async_to_lds_b128 %0, %1, %2"
                   :: "v"(ldsOff + j * 16u), "v"(gOff + j * 16u), "s"(wtBase)
                   : "memory");
  };
  auto mmaStage = [&](const _Float16* aB, const _Float16* bB) {
    v16h afr[4];
    for (int mt = 0; mt < 4; ++mt)
      afr[mt] = load_frag_a(aB, 32, waveM * 64 + mt * 16, 0, lane);
    for (int nt = 0; nt < 4; ++nt) {
      v16h bfr = load_frag_b(bB, 32, waveN * 64 + nt * 16, 0, lane);
      for (int mt = 0; mt < 4; ++mt)
        acc[mt][nt] = __builtin_amdgcn_wmma_f32_16x16x32_f16(
            false, afr[mt], false, bfr, (short)0, acc[mt][nt], false, false);
    }
  };

  asyncB(bOff0, 0);
  loadA(0);
  for (int k0 = 0; k0 < D_IN; k0 += 2 * K_CHUNK) {
    // ---- phase 0: chunk k0 in buffers 0 ----
    storeA(aBuf0);
    asm volatile("s_wait_asynccnt 0x0" ::: "memory");
    __syncthreads();
    asyncB(bOff1, k0 + K_CHUNK);      // k0+32 < 1024 always (k0 <= 960)
    loadA(k0 + K_CHUNK);
    mmaStage(aBuf0, bBuf0);
    // ---- phase 1: chunk k0+32 in buffers 1 ----
    storeA(aBuf1);
    asm volatile("s_wait_asynccnt 0x0" ::: "memory");
    __syncthreads();
    if (k0 + 2 * K_CHUNK < D_IN) {
      asyncB(bOff0, k0 + 2 * K_CHUNK);
      loadA(k0 + 2 * K_CHUNK);
    }
    mmaStage(aBuf1, bBuf1);
  }
  __syncthreads();   // tiles dead; smem becomes bufP

  // ====== epilogue: bias + sigmoid + reg partial (f32 in regs) + store P =====
  float regAcc = 0.0f;
  for (int nt = 0; nt < 4; ++nt) {
    int c = waveN * 64 + nt * 16 + l16;
    float bb = 0.0f, scale = 0.0f;
    if (c < N_REAL) {
      bb = b_split[c];
      int d = 31 - __clz(c + 1);                       // depth of node c
      scale = -0.5f * ldexpf(1.0f, -d) * (1.0f / (float)B_ROWS);
    }
    for (int mt = 0; mt < 4; ++mt) {
      int rloc = waveM * 64 + mt * 16 + 8 * half;
      for (int i = 0; i < 8; ++i) {
        float z = acc[mt][nt][i] + bb;
        float p = __builtin_amdgcn_rcpf(1.0f + __expf(-z));   // sigmoid, v_rcp+v_exp
        float q = fmaxf(p * (1.0f - p), 1e-5f);
        regAcc += scale * __logf(q);
        bufP[(size_t)(rloc + i) * N_NODES + c] = (_Float16)p;
      }
    }
  }
  __syncthreads();

  // ========== tree expansion: 256 leaf probs per row, sibling sharing =========
  for (int it = 0; it < 8; ++it) {
    int pair = tid + it * BLOCK;        // 0..2047 = (row, high-nibble)
    int row = pair >> 4, h = pair & 15;
    const _Float16* Pr = bufP + (size_t)row * N_NODES;
    float common = 1.0f;
    for (int d = 0; d < 4; ++d) {
      int node = (1 << d) - 1 + (h >> (4 - d));
      int bit  = (h >> (3 - d)) & 1;
      float p = (float)Pr[node];
      common *= bit ? p : (1.0f - p);
    }
    float p4 = (float)Pr[15 + h];
    float p5[2], p6[4], p7[8];
    for (int j = 0; j < 2; ++j) p5[j] = (float)Pr[31 + 2 * h + j];
    for (int j = 0; j < 4; ++j) p6[j] = (float)Pr[63 + 4 * h + j];
    for (int j = 0; j < 8; ++j) p7[j] = (float)Pr[127 + 8 * h + j];
    _Float16* Qr = bufQ + (size_t)row * N_NODES + h * 16;
    for (int v = 0; v < 16; ++v) {
      float pr = common;
      pr *= ((v >> 3) & 1) ? p4 : (1.0f - p4);
      float q5 = p5[v >> 3]; pr *= ((v >> 2) & 1) ? q5 : (1.0f - q5);
      float q6 = p6[v >> 2]; pr *= ((v >> 1) & 1) ? q6 : (1.0f - q6);
      float q7 = p7[v >> 1]; pr *= (v & 1)        ? q7 : (1.0f - q7);
      Qr[v] = (_Float16)pr;
    }
  }
  __syncthreads();

  // ============ GEMM2: out[128][128] = probs[128][256] @ leafT^T ============
  const int waveM2 = wave >> 1;   // 0..3 -> 32-row band
  const int waveN2 = wave & 1;    // 0..1 -> 64-col band
  v8f acc2[2][4];
  for (int mt = 0; mt < 2; ++mt)
    for (int nt = 0; nt < 4; ++nt) acc2[mt][nt] = vzero;

  for (int k0 = 0; k0 < N_NODES; k0 += K_CHUNK) {
    v16h afr[2];
    for (int mt = 0; mt < 2; ++mt)
      afr[mt] = load_frag_a(bufQ, N_NODES, waveM2 * 32 + mt * 16, k0, lane);
    for (int nt = 0; nt < 4; ++nt) {
      v16h bfr = load_frag_b(leafT, N_NODES, waveN2 * 64 + nt * 16, k0, lane);  // L2-resident
      for (int mt = 0; mt < 2; ++mt)
        acc2[mt][nt] = __builtin_amdgcn_wmma_f32_16x16x32_f16(
            false, afr[mt], false, bfr, (short)0, acc2[mt][nt], false, false);
    }
  }
  for (int mt = 0; mt < 2; ++mt) {
    int rloc = waveM2 * 32 + mt * 16 + 8 * half;
    for (int nt = 0; nt < 4; ++nt) {
      int c = waveN2 * 64 + nt * 16 + l16;
      for (int i = 0; i < 8; ++i)
        out[(size_t)(rowBase + rloc + i) * D_OUT + c] = acc2[mt][nt][i];
    }
  }

  // ================= deterministic reg reduction (block level) ==============
  __syncthreads();
  float* sred = (float*)smem;
  sred[tid] = regAcc;
  __syncthreads();
  for (int s = BLOCK / 2; s > 0; s >>= 1) {
    if (tid < s) sred[tid] += sred[tid + s];
    __syncthreads();
  }
  if (tid == 0) regPartial[blockIdx.x] = sred[0];
}

__global__ void reduce_reg(const float* __restrict__ part, float* __restrict__ outreg) {
  __shared__ float s[256];
  int tid = threadIdx.x;
  s[tid] = part[tid] + part[tid + 256];
  __syncthreads();
  for (int st = 128; st > 0; st >>= 1) {
    if (tid < st) s[tid] += s[tid + st];
    __syncthreads();
  }
  if (tid == 0) *outreg = s[0];
}

extern "C" void kernel_launch(void* const* d_in, const int* in_sizes, int n_in,
                              void* d_out, int out_size, void* d_ws, size_t ws_size,
                              hipStream_t stream) {
  const float* x = (const float*)d_in[0];       // (65536, 1024)
  const float* W = (const float*)d_in[1];       // (1024, 255)
  const float* b = (const float*)d_in[2];       // (255,)
  const float* L = (const float*)d_in[3];       // (256, 128)
  float* out = (float*)d_out;                   // 65536*128 floats + 1 reg scalar

  char* ws = (char*)d_ws;
  _Float16* Wt      = (_Float16*)ws;                            // 512 KB
  _Float16* leafT   = (_Float16*)(ws + 512 * 1024);             // 64 KB
  float*    regPart = (float*)   (ws + 512 * 1024 + 64 * 1024); // 2 KB

  prep_wt  <<<1024, BLOCK, 0, stream>>>(W, Wt);
  prep_leaf<<<128,  BLOCK, 0, stream>>>(L, leafT);
  fused_tree_kernel<<<N_BLOCKS, BLOCK, 0, stream>>>(x, b, Wt, leafT, out, regPart);
  reduce_reg<<<1, BLOCK, 0, stream>>>(regPart, out + (size_t)B_ROWS * D_OUT);
}
